// DiffusionModule_67259187855466
// MI455X (gfx1250) — matrix operand order repaired and checked by reading, not compiled, not run on hardware
//
#include <hip/hip_runtime.h>
#include <math.h>

// ---------------------------------------------------------------------------
// MI455X (gfx1250) diffusion-transformer implementation.
// wave32, WMMA 16x16x32 f16->f32. Memory-bound on the pair tensor: we
// contract pair [L,L,64] against all NB*H=32 head weights ONCE, streaming the
// 1 GiB pair tensor through LDS with the Tensor Data Mover (double-buffered
// tensor_load_to_lds + s_wait_tensorcnt), store f16 bias planes [32][L][L],
// then run flash-attention per head. Total pair traffic ~1.5 GiB -> ~65us at
// 23.3 TB/s, which dominates; all GEMMs (~30 GFLOP f16 WMMA) are far cheaper.
// ---------------------------------------------------------------------------

#define L_TOK 2048
#define C_ATOM 256
#define NBLK 4
#define NHEAD 8
#define HDIM 32

typedef __attribute__((ext_vector_type(16))) _Float16 v16h;
typedef __attribute__((ext_vector_type(8)))  float    v8f;
typedef __attribute__((ext_vector_type(4)))  unsigned int u32x4;
typedef __attribute__((ext_vector_type(8)))  unsigned int u32x8;

__device__ __forceinline__ v8f wmma16x16x32(v16h a, v16h b, v8f c) {
  return __builtin_amdgcn_wmma_f32_16x16x32_f16(false, a, false, b, (short)0, c,
                                                false, false);
}

__device__ __forceinline__ float gelu_exact(float x) {
  return 0.5f * x * (1.0f + erff(x * 0.70710678118654752f));
}

// reductions across the 16 lanes that hold one C-tile row (wave32, two groups)
__device__ __forceinline__ float rowmax16(float x) {
  for (int off = 8; off; off >>= 1) x = fmaxf(x, __shfl_xor(x, off, 16));
  return x;
}
__device__ __forceinline__ float rowsum16(float x) {
  for (int off = 8; off; off >>= 1) x += __shfl_xor(x, off, 16);
  return x;
}

// A-matrix (f16 16x32) per-lane K index for half j:
// lanes 0-15: K = {0..15}; lanes 16-31: K = {16..31}
__device__ __forceinline__ int a_kidx(int j, int lane) {
  return ((lane < 16) ? 0 : 16) + j;
}

// low 32 bits of a generic pointer to LDS == LDS byte offset (aperture map)
__device__ __forceinline__ unsigned lds_off(const void* p) {
  return (unsigned)(uintptr_t)p;
}

// ---------------------------------------------------------------------------
// Generic GEMM: C[M,N] = act(A[M,K] @ W[N,K]^T + bias[N]) (+ R[M,N])
// One wave per 16x16 tile; K multiple of 32.
// ---------------------------------------------------------------------------
__global__ __launch_bounds__(32) void wmma_gemm_kernel(
    const float* __restrict__ A, const float* __restrict__ W,
    const float* __restrict__ bias, const float* __restrict__ R,
    float* __restrict__ C, int M, int N, int K, int act) {
  const int n0 = blockIdx.x * 16;
  const int m0 = blockIdx.y * 16;
  const int l = threadIdx.x;
  const int lane16 = l & 15;
  const int boff = (l < 16) ? 0 : 16;
  const float* __restrict__ arow = A + (size_t)(m0 + lane16) * K;
  const float* __restrict__ wrow = W + (size_t)(n0 + lane16) * K;
  v8f acc = {};
  for (int kt = 0; kt < K; kt += 32) {
    __builtin_prefetch(arow + kt + 64, 0, 1);
    v16h a, b;
#pragma unroll
    for (int j = 0; j < 16; ++j) {
      a[j] = (_Float16)arow[kt + a_kidx(j, l)];
      b[j] = (_Float16)wrow[kt + boff + j];
    }
    acc = wmma16x16x32(a, b, acc);
  }
  const int mg = (l >> 4) * 8;
#pragma unroll
  for (int r = 0; r < 8; ++r) {
    const int m = m0 + mg + r;
    const int n = n0 + lane16;
    float vv = acc[r];
    if (bias) vv += bias[n];
    if (act == 1) vv = gelu_exact(vv);
    if (R) vv += R[(size_t)m * N + n];
    C[(size_t)m * N + n] = vv;
  }
}

// ---------------------------------------------------------------------------
// Pair-bias contraction: biasF16[c][i*L+j] = sum_ch pair[i,j,ch] * pWall[c,ch]
// viewed as GEMM [L*L, 64] x [64, 32]; c = blk*8 + head. Output f16.
// The pair stream (1 GiB) is staged into LDS by the Tensor Data Mover with
// double buffering: 64x64-f32 tiles (16 KB), TENSORcnt-tracked.
// ---------------------------------------------------------------------------
#define PB_ROWS 64    // rows of [L*L,64] per TDM tile
#define PB_CHUNKS 4   // tiles per workgroup

__global__ __launch_bounds__(32) void pair_bias_kernel(
    const float* __restrict__ P, const float* __restrict__ pW,
    _Float16* __restrict__ B) {
  __shared__ float tile[2][PB_ROWS * 64];
  const size_t LL = (size_t)L_TOK * L_TOK;
  const int l = threadIdx.x;
  const int lane16 = l & 15;
  const int boff = (l < 16) ? 0 : 16;
  const size_t row0 = (size_t)blockIdx.x * (PB_ROWS * PB_CHUNKS);

  // hoisted B tiles: pWall is [32,64], reused for every chunk
  v16h b0[2], b1[2];
#pragma unroll
  for (int kc = 0; kc < 2; ++kc)
#pragma unroll
    for (int j = 0; j < 16; ++j) {
      b0[kc][j] = (_Float16)pW[(size_t)lane16 * 64 + kc * 32 + boff + j];
      b1[kc][j] = (_Float16)pW[(size_t)(16 + lane16) * 64 + kc * 32 + boff + j];
    }

  // Build + issue a TDM descriptor (D#) for one 64x64-f32 tile -> LDS buf.
  auto tdm_issue = [&](int chunk, int buf) {
    const unsigned lds = lds_off(&tile[buf][0]);
    const unsigned long long ga = (unsigned long long)(uintptr_t)(
        P + (row0 + (size_t)chunk * PB_ROWS) * 64);
    u32x4 g0;
    g0[0] = 1u;                                   // count=1, user descriptor
    g0[1] = lds;                                  // lds_addr (bytes)
    g0[2] = (unsigned)(ga & 0xFFFFFFFFu);         // global_addr[31:0]
    g0[3] = (unsigned)((ga >> 32) & 0x01FFFFFFu)  // global_addr[56:32]
            | (2u << 30);                         // type = 2 ("image")
    u32x8 g1;
    g1[0] = 2u << 16;            // data_size = 4 bytes; no pad/iter/multicast
    g1[1] = 64u << 16;           // tensor_dim0 = 64 elements
    g1[2] = 0u;                  // tensor_dim0 hi / tensor_dim1 lo16 (0)
    g1[3] = 0x40u | (64u << 16); // tensor_dim1 = 0x400000 (hi16) ; tile_dim0=64
    g1[4] = (unsigned)PB_ROWS;   // tile_dim1 = 64, tile_dim2 = 0
    g1[5] = 64u;                 // tensor_dim0_stride = 64
    g1[6] = 0u;                  // stride hi / dim1_stride lo (unused, 2D)
    g1[7] = 0u;
    asm volatile("tensor_load_to_lds %0, %1" ::"s"(g0), "s"(g1) : "memory");
  };

  tdm_issue(0, 0);
  for (int c = 0; c < PB_CHUNKS; ++c) {
    if (c + 1 < PB_CHUNKS) {
      tdm_issue(c + 1, (c + 1) & 1);
      __builtin_amdgcn_s_wait_tensorcnt(1);  // oldest (chunk c) done
    } else {
      __builtin_amdgcn_s_wait_tensorcnt(0);
    }
    const float* __restrict__ tb = &tile[c & 1][0];
#pragma unroll
    for (int rt = 0; rt < PB_ROWS / 16; ++rt) {
      v8f acc0 = {}, acc1 = {};
#pragma unroll
      for (int kc = 0; kc < 2; ++kc) {
        v16h a;
#pragma unroll
        for (int j = 0; j < 16; ++j)
          a[j] = (_Float16)tb[(rt * 16 + lane16) * 64 + kc * 32 + a_kidx(j, l)];
        acc0 = wmma16x16x32(a, b0[kc], acc0);
        acc1 = wmma16x16x32(a, b1[kc], acc1);
      }
      const size_t mbase =
          row0 + (size_t)c * PB_ROWS + rt * 16 + (size_t)((l >> 4) * 8);
#pragma unroll
      for (int r = 0; r < 8; ++r) {
        B[(size_t)lane16 * LL + mbase + r]        = (_Float16)acc0[r];
        B[(size_t)(16 + lane16) * LL + mbase + r] = (_Float16)acc1[r];
      }
    }
  }
}

// ---------------------------------------------------------------------------
// Flash attention, one wave per (head, 16-query tile). d=32 -> K-dim of one
// WMMA. Online softmax; P re-laid out C-tile -> A-tile through LDS.
// ---------------------------------------------------------------------------
__global__ __launch_bounds__(32) void attn_kernel(
    const float* __restrict__ Q, const float* __restrict__ Km,
    const float* __restrict__ V, const _Float16* __restrict__ biasF,
    float* __restrict__ O, int hbase) {
  __shared__ float sP[16 * 32];
  const int q0 = blockIdx.x * 16;
  const int h = blockIdx.y;
  const int l = threadIdx.x;
  const int lane16 = l & 15;
  const int boff = (l < 16) ? 0 : 16;
  const int hc = h * HDIM;
  const int mg = (l >> 4) * 8;
  const _Float16* __restrict__ bpl =
      biasF + (size_t)(hbase + h) * L_TOK * L_TOK;

  // Q tile, pre-scaled by 1/sqrt(d)
  v16h aq;
#pragma unroll
  for (int j = 0; j < 16; ++j)
    aq[j] = (_Float16)(Q[(size_t)(q0 + lane16) * C_ATOM + hc + a_kidx(j, l)] *
                       0.17677669529663687f);

  float mrun[8], lrun[8];
  v8f o0 = {}, o1 = {};
#pragma unroll
  for (int r = 0; r < 8; ++r) { mrun[r] = -1e30f; lrun[r] = 0.0f; }

  for (int k0 = 0; k0 < L_TOK; k0 += 32) {
    // K^T tiles for S = Q K^T  (B element (kdim, key))
    v16h bk0, bk1;
#pragma unroll
    for (int j = 0; j < 16; ++j) {
      bk0[j] = (_Float16)Km[(size_t)(k0 + lane16) * C_ATOM + hc + boff + j];
      bk1[j] = (_Float16)Km[(size_t)(k0 + 16 + lane16) * C_ATOM + hc + boff + j];
    }
    v8f z = {};
    v8f s0 = wmma16x16x32(aq, bk0, z);
    v8f s1 = wmma16x16x32(aq, bk1, z);

    // bias + online softmax (row M = r + mg lives in 16 lanes of one group)
#pragma unroll
    for (int r = 0; r < 8; ++r) {
      const int m = q0 + mg + r;
      const size_t ro = (size_t)m * L_TOK + k0;
      float t0 = s0[r] + (float)bpl[ro + lane16];
      float t1 = s1[r] + (float)bpl[ro + 16 + lane16];
      float mnew = fmaxf(mrun[r], rowmax16(fmaxf(t0, t1)));
      float al = __expf(mrun[r] - mnew);
      float e0 = __expf(t0 - mnew);
      float e1 = __expf(t1 - mnew);
      lrun[r] = al * lrun[r] + rowsum16(e0 + e1);
      mrun[r] = mnew;
      o0[r] *= al;
      o1[r] *= al;
      sP[(mg + r) * 32 + lane16] = e0;
      sP[(mg + r) * 32 + 16 + lane16] = e1;
    }
    __syncthreads();

    // P (16 q x 32 keys) as A; V chunk (32 keys x 32 dims) as two B tiles
    v16h ap, bv0, bv1;
#pragma unroll
    for (int j = 0; j < 16; ++j)
      ap[j] = (_Float16)sP[lane16 * 32 + a_kidx(j, l)];
#pragma unroll
    for (int j = 0; j < 16; ++j) {
      const float* vr = V + (size_t)(k0 + boff + j) * C_ATOM + hc;
      bv0[j] = (_Float16)vr[lane16];
      bv1[j] = (_Float16)vr[16 + lane16];
    }
    o0 = wmma16x16x32(ap, bv0, o0);
    o1 = wmma16x16x32(ap, bv1, o1);
    __syncthreads();
  }

#pragma unroll
  for (int r = 0; r < 8; ++r) {
    const int m = q0 + mg + r;
    const float inv = 1.0f / lrun[r];
    O[(size_t)m * C_ATOM + hc + lane16]      = o0[r] * inv;
    O[(size_t)m * C_ATOM + hc + 16 + lane16] = o1[r] * inv;
  }
}

// ---------------------------------------------------------------------------
// adaLN: y = LN(x)*g+b, then *(1+scale)+shift with ss=[scale(256),shift(256)]
// ---------------------------------------------------------------------------
__global__ __launch_bounds__(256) void adaln_kernel(
    const float* __restrict__ X, const float* __restrict__ g,
    const float* __restrict__ b, const float* __restrict__ ss,
    float* __restrict__ Y) {
  __shared__ float red[256];
  const int row = blockIdx.x, t = threadIdx.x;
  const float v = X[(size_t)row * C_ATOM + t];
  red[t] = v;
  __syncthreads();
  for (int s = 128; s; s >>= 1) {
    if (t < s) red[t] += red[t + s];
    __syncthreads();
  }
  const float mu = red[0] * (1.0f / 256.0f);
  __syncthreads();
  const float d = v - mu;
  red[t] = d * d;
  __syncthreads();
  for (int s = 128; s; s >>= 1) {
    if (t < s) red[t] += red[t + s];
    __syncthreads();
  }
  const float var = red[0] * (1.0f / 256.0f);
  const float y = d * rsqrtf(var + 1e-5f) * g[t] + b[t];
  Y[(size_t)row * C_ATOM + t] = y * (1.0f + ss[t]) + ss[256 + t];
}

// ---------------------------------------------------------------------------
// Time embedding MLP + all adaLN conditioning vectors (tiny; one block)
// ---------------------------------------------------------------------------
__global__ __launch_bounds__(256) void time_embed_kernel(
    const float* __restrict__ sigma, const float* __restrict__ W1,
    const float* __restrict__ b1, const float* __restrict__ W2,
    const float* __restrict__ b2, const float* __restrict__ a1pW,
    const float* __restrict__ a1pb, const float* __restrict__ a2pW,
    const float* __restrict__ a2pb, float* __restrict__ tc,
    float* __restrict__ ss1, float* __restrict__ ss2) {
  __shared__ float temb[256];
  __shared__ float hmid[1024];
  __shared__ float tcs[256];
  const int t = threadIdx.x;
  const float cn = 0.25f * logf(sigma[0] + 1e-8f);
  {
    const int i = (t < 128) ? t : (t - 128);
    const float fr = __expf(-logf(10000.0f) * (float)i / 128.0f);
    temb[t] = (t < 128) ? cosf(cn * fr) : sinf(cn * fr);
  }
  __syncthreads();
  for (int p = 0; p < 4; ++p) {
    const int oi = t + 256 * p;
    float s = b1[oi];
    const float* wr = W1 + (size_t)oi * 256;
    for (int c = 0; c < 256; ++c) s += temb[c] * wr[c];
    hmid[oi] = gelu_exact(s);
  }
  __syncthreads();
  {
    float s = b2[t];
    const float* wr = W2 + (size_t)t * 1024;
    for (int c = 0; c < 1024; ++c) s += hmid[c] * wr[c];
    tcs[t] = s;
    tc[t] = s;
  }
  __syncthreads();
  for (int blk = 0; blk < NBLK; ++blk)
    for (int p = 0; p < 2; ++p) {
      const int oi = t + 256 * p;
      float s1 = a1pb[blk * 512 + oi];
      float s2 = a2pb[blk * 512 + oi];
      const float* w1 = a1pW + ((size_t)blk * 512 + oi) * 256;
      const float* w2 = a2pW + ((size_t)blk * 512 + oi) * 256;
      for (int c = 0; c < 256; ++c) {
        s1 += tcs[c] * w1[c];
        s2 += tcs[c] * w2[c];
      }
      ss1[blk * 512 + oi] = s1;
      ss2[blk * 512 + oi] = s2;
    }
}

// h += c_in * x_noisy @ coord_W^T + coord_b   (K=3, not WMMA-worthy)
__global__ __launch_bounds__(256) void coord_add_kernel(
    const float* __restrict__ x, const float* __restrict__ sigma,
    const float* __restrict__ cW, const float* __restrict__ cb,
    float* __restrict__ h) {
  const int m = blockIdx.x, n = threadIdx.x;
  const float sg = sigma[0];
  const float cin = rsqrtf(sg * sg + 256.0f);
  const float* xr = x + (size_t)m * 3;
  h[(size_t)m * C_ATOM + n] +=
      cin * (xr[0] * cW[n * 3] + xr[1] * cW[n * 3 + 1] + xr[2] * cW[n * 3 + 2]) +
      cb[n];
}

// out = c_skip*x + c_out*(h @ out_W^T + out_b)
__global__ __launch_bounds__(256) void final_kernel(
    const float* __restrict__ h, const float* __restrict__ x,
    const float* __restrict__ sigma, const float* __restrict__ oW,
    const float* __restrict__ ob, float* __restrict__ out) {
  const int idx = blockIdx.x * blockDim.x + threadIdx.x;
  if (idx >= L_TOK * 3) return;
  const int m = idx / 3, j = idx % 3;
  const float sg = sigma[0];
  const float s2 = sg * sg + 256.0f;
  const float cskip = 256.0f / s2;
  const float cout = sg * 16.0f * rsqrtf(s2);
  float s = ob[j];
  const float* hr = h + (size_t)m * C_ATOM;
  const float* wr = oW + (size_t)j * C_ATOM;
  for (int c = 0; c < 256; ++c) s += hr[c] * wr[c];
  out[idx] = cskip * x[idx] + cout * s;
}

// ---------------------------------------------------------------------------
extern "C" void kernel_launch(void* const* d_in, const int* in_sizes, int n_in,
                              void* d_out, int out_size, void* d_ws,
                              size_t ws_size, hipStream_t stream) {
  (void)in_sizes; (void)n_in; (void)out_size; (void)ws_size;
  const float* x_noisy  = (const float*)d_in[0];
  const float* sigma    = (const float*)d_in[1];
  const float* single   = (const float*)d_in[2];
  const float* pair     = (const float*)d_in[3];
  const float* coord_W  = (const float*)d_in[4];
  const float* coord_b  = (const float*)d_in[5];
  const float* single_W = (const float*)d_in[6];
  const float* single_b = (const float*)d_in[7];
  const float* tmlp_W1  = (const float*)d_in[8];
  const float* tmlp_b1  = (const float*)d_in[9];
  const float* tmlp_W2  = (const float*)d_in[10];
  const float* tmlp_b2  = (const float*)d_in[11];
  const float* ada1_g   = (const float*)d_in[12];
  const float* ada1_b   = (const float*)d_in[13];
  const float* ada1_pW  = (const float*)d_in[14];
  const float* ada1_pb  = (const float*)d_in[15];
  const float* qW       = (const float*)d_in[16];
  const float* kW       = (const float*)d_in[17];
  const float* vW       = (const float*)d_in[18];
  const float* pairW    = (const float*)d_in[19];
  const float* outW     = (const float*)d_in[20];
  const float* outb     = (const float*)d_in[21];
  const float* ada2_g   = (const float*)d_in[22];
  const float* ada2_b   = (const float*)d_in[23];
  const float* ada2_pW  = (const float*)d_in[24];
  const float* ada2_pb  = (const float*)d_in[25];
  const float* ffn_W1   = (const float*)d_in[26];
  const float* ffn_b1   = (const float*)d_in[27];
  const float* ffn_W2   = (const float*)d_in[28];
  const float* ffn_b2   = (const float*)d_in[29];
  const float* out_W    = (const float*)d_in[30];
  const float* out_b    = (const float*)d_in[31];
  float* out = (float*)d_out;

  const size_t LC = (size_t)L_TOK * C_ATOM;
  float* ws = (float*)d_ws;
  size_t off = 0;
  float* tc  = ws + off; off += 256;
  float* ss1 = ws + off; off += NBLK * 512;
  float* ss2 = ws + off; off += NBLK * 512;
  float* h   = ws + off; off += LC;
  float* h1  = ws + off; off += LC;
  float* q   = ws + off; off += LC;
  float* k   = ws + off; off += LC;
  float* v   = ws + off; off += LC;
  float* o   = ws + off; off += LC;
  float* hff = ws + off; off += (size_t)L_TOK * 1024;
  _Float16* biasF = (_Float16*)(ws + off);  // 32 * L * L halves (256 MiB)

  const dim3 wave(32);

  // 1) time embedding + all conditioning vectors
  time_embed_kernel<<<1, 256, 0, stream>>>(sigma, tmlp_W1, tmlp_b1, tmlp_W2,
                                           tmlp_b2, ada1_pW, ada1_pb, ada2_pW,
                                           ada2_pb, tc, ss1, ss2);
  // 2) h = single @ single_W^T + single_b ; then += coord embedding
  wmma_gemm_kernel<<<dim3(C_ATOM / 16, L_TOK / 16), wave, 0, stream>>>(
      single, single_W, single_b, nullptr, h, L_TOK, C_ATOM, C_ATOM, 0);
  coord_add_kernel<<<L_TOK, 256, 0, stream>>>(x_noisy, sigma, coord_W, coord_b,
                                              h);
  // 3) pair bias planes for all blocks/heads in one TDM-streamed pass
  pair_bias_kernel<<<(unsigned)((size_t)L_TOK * L_TOK / (PB_ROWS * PB_CHUNKS)),
                     wave, 0, stream>>>(pair, pairW, biasF);

  for (int blk = 0; blk < NBLK; ++blk) {
    const float* qWb = qW + (size_t)blk * C_ATOM * C_ATOM;
    const float* kWb = kW + (size_t)blk * C_ATOM * C_ATOM;
    const float* vWb = vW + (size_t)blk * C_ATOM * C_ATOM;
    const float* oWb = outW + (size_t)blk * C_ATOM * C_ATOM;
    const dim3 g256(C_ATOM / 16, L_TOK / 16);

    adaln_kernel<<<L_TOK, 256, 0, stream>>>(h, ada1_g + blk * 256,
                                            ada1_b + blk * 256,
                                            ss1 + blk * 512, h1);
    wmma_gemm_kernel<<<g256, wave, 0, stream>>>(h1, qWb, nullptr, nullptr, q,
                                                L_TOK, C_ATOM, C_ATOM, 0);
    wmma_gemm_kernel<<<g256, wave, 0, stream>>>(h1, kWb, nullptr, nullptr, k,
                                                L_TOK, C_ATOM, C_ATOM, 0);
    wmma_gemm_kernel<<<g256, wave, 0, stream>>>(h1, vWb, nullptr, nullptr, v,
                                                L_TOK, C_ATOM, C_ATOM, 0);
    attn_kernel<<<dim3(L_TOK / 16, NHEAD), wave, 0, stream>>>(q, k, v, biasF, o,
                                                              blk * NHEAD);
    // h = h + o @ outW^T + outb (in-place residual: element-wise read->write)
    wmma_gemm_kernel<<<g256, wave, 0, stream>>>(o, oWb, outb + blk * 256, h, h,
                                                L_TOK, C_ATOM, C_ATOM, 0);
    adaln_kernel<<<L_TOK, 256, 0, stream>>>(h, ada2_g + blk * 256,
                                            ada2_b + blk * 256,
                                            ss2 + blk * 512, h1);
    wmma_gemm_kernel<<<dim3(1024 / 16, L_TOK / 16), wave, 0, stream>>>(
        h1, ffn_W1 + (size_t)blk * 1024 * 256, ffn_b1 + blk * 1024, nullptr,
        hff, L_TOK, 1024, C_ATOM, 1);
    wmma_gemm_kernel<<<g256, wave, 0, stream>>>(
        hff, ffn_W2 + (size_t)blk * 256 * 1024, ffn_b2 + blk * 256, h, h,
        L_TOK, C_ATOM, 1024, 0);
  }

  final_kernel<<<(L_TOK * 3 + 255) / 256, 256, 0, stream>>>(h, x_noisy, sigma,
                                                            out_W, out_b, out);
}
